// ExponentialRNN_12292196401864
// MI455X (gfx1250) — matrix operand order repaired and checked by reading, not compiled, not run on hardware
//
#include <hip/hip_runtime.h>
#include <hip/hip_bf16.h>

// ExponentialRNN for MI455X (gfx1250, wave32, WMMA).
//
// Phase 0: pack T (256x512) and B (512x512) fp32 -> bf16 in exact
//          v_wmma_f32_16x16x32_bf16 B-operand per-lane layout (32B/lane slots).
// Phase 1: xT = x @ T  (M=32768,K=256,N=512) bf16 WMMA GEMM -> ws ([t,b,u] fp32).
// Phase 2: 512-step recurrent scan, 4 workgroups (batch tiles of 16).
//          B K-low half register-resident (128 VGPR/wave), K-high half in LDS
//          (256KB), h state ping-pong in LDS in A-operand layout (2x16KB).
//
// Workspace requirement: 64MB (xT) + 256KB (Tpack) + 512KB (Bpack) ~= 64.75MB.

typedef __attribute__((ext_vector_type(16))) __bf16 v16bf;
typedef __attribute__((ext_vector_type(8)))  float  v8f;

#define BATCH 64
#define SEQ   512
#define DIN   256
#define UNITS 512

#define NTILES 32          // UNITS/16 output column tiles
#define KCH_B  16          // UNITS/32 K-chunks for h@B
#define KCH_T  8           // DIN/32   K-chunks for x@T

#define XT_BYTES    ((size_t)SEQ * BATCH * UNITS * 4)   // 67108864
#define TPACK_OFF   XT_BYTES
#define TPACK_SLOTS (NTILES * KCH_T * 32)               // 8192 x 32B
#define BPACK_OFF   (TPACK_OFF + (size_t)TPACK_SLOTS * 32)
#define BPACK_SLOTS (NTILES * KCH_B * 32)               // 16384 x 32B

// LDS layout for the scan kernel
#define LDS_BH_BYTES   (NTILES * 8 * 32 * 32)           // 262144: B K-high half
#define LDS_H_BYTES    (KCH_B * 32 * 32)                // 16384 per h buffer
#define LDS_TOTAL      (LDS_BH_BYTES + 2 * LDS_H_BYTES) // 294912

__device__ __forceinline__ unsigned short f2bf_bits(float f) {
    __bf16 b = (__bf16)f;
    return __builtin_bit_cast(unsigned short, b);
}

// ---------------------------------------------------------------------------
// Phase 0: pack T and B into WMMA B-operand layout.
// B-operand (32x16 KxN bf16, ISA 7.12.4 striping): lane = (g<<4)|n,
//   word j (0..7) holds K = c*32 + (j>>2)*16 + g*8 + (j&3)*2 (+1), column n.
// Slot layout: [ntile][chunk][lane] * 32B.
// ---------------------------------------------------------------------------
__global__ void __launch_bounds__(256) pack_weights(
        const float* __restrict__ Tm, const float* __restrict__ Bm,
        char* __restrict__ ws) {
    int tid = blockIdx.x * blockDim.x + threadIdx.x;
    const float* src;
    char* dst;
    int nchunks, slot;
    if (tid < TPACK_SLOTS) {
        src = Tm; dst = ws + TPACK_OFF; nchunks = KCH_T; slot = tid;
    } else if (tid < TPACK_SLOTS + BPACK_SLOTS) {
        src = Bm; dst = ws + BPACK_OFF; nchunks = KCH_B; slot = tid - TPACK_SLOTS;
    } else {
        return;
    }
    int lane = slot & 31;
    int sc   = slot >> 5;
    int c    = sc % nchunks;
    int nt   = sc / nchunks;
    int n    = nt * 16 + (lane & 15);
    int g    = lane >> 4;
    v16bf v;
#pragma unroll
    for (int j = 0; j < 8; ++j) {
        int k = c * 32 + (j >> 2) * 16 + g * 8 + (j & 3) * 2;
        v[2 * j]     = (__bf16)src[(size_t)k * UNITS + n];
        v[2 * j + 1] = (__bf16)src[(size_t)(k + 1) * UNITS + n];
    }
    *(v16bf*)(dst + (size_t)slot * 32) = v;
}

// ---------------------------------------------------------------------------
// Phase 1: xT[t][b][u] = sum_d x[b][t][d] * T[d][u]
// One block = one 16-row M tile (rows = b*SEQ + t, never straddles b).
// 8 waves x 4 N-tiles; A operand built from fp32 x via float4 loads + cvt.
// A-operand layout (16-bit A 16x32, ISA 7.12.2): lane=(g<<4)|m,
//   word j: K = (j>>2)*16 + g*8 + (j&3)*2 (+1).
// ---------------------------------------------------------------------------
__global__ void __launch_bounds__(256) proj_gemm(
        const float* __restrict__ x, const char* __restrict__ tpack,
        float* __restrict__ xT) {
    int mtile = blockIdx.x;                 // 0..2047
    int wave  = threadIdx.x >> 5;           // 0..7
    int lane  = threadIdx.x & 31;
    int m = lane & 15, g = lane >> 4;
    int rb = mtile * 16;
    int b  = rb >> 9;                       // rb / SEQ
    int t0 = rb & (SEQ - 1);
    int nt0 = wave * 4;

    const float* xrow = x + (size_t)(rb + m) * DIN;
    v8f acc[4] = {};

#pragma unroll
    for (int c = 0; c < KCH_T; ++c) {
        v16bf a;
#pragma unroll
        for (int blk = 0; blk < 2; ++blk) {
            float4 f0 = *(const float4*)(xrow + c * 32 + blk * 16 + g * 8);
            float4 f1 = *(const float4*)(xrow + c * 32 + blk * 16 + g * 8 + 4);
            int j = blk * 4;
            a[2*(j+0)] = (__bf16)f0.x; a[2*(j+0)+1] = (__bf16)f0.y;
            a[2*(j+1)] = (__bf16)f0.z; a[2*(j+1)+1] = (__bf16)f0.w;
            a[2*(j+2)] = (__bf16)f1.x; a[2*(j+2)+1] = (__bf16)f1.y;
            a[2*(j+3)] = (__bf16)f1.z; a[2*(j+3)+1] = (__bf16)f1.w;
        }
#pragma unroll
        for (int i = 0; i < 4; ++i) {
            v16bf bop = *(const v16bf*)(tpack +
                (size_t)(((nt0 + i) * KCH_T + c) * 32 + lane) * 32);
            acc[i] = __builtin_amdgcn_wmma_f32_16x16x32_bf16(
                false, a, false, bop, (short)0, acc[i], false, false);
        }
    }
    // C layout: lane holds column n = lane&15, rows r + 8*g. Store [t,b,u].
    int n = lane & 15;
#pragma unroll
    for (int i = 0; i < 4; ++i) {
        int u = (nt0 + i) * 16 + n;
#pragma unroll
        for (int r = 0; r < 8; ++r) {
            int mr = r + 8 * g;
            xT[(size_t)((t0 + mr) * BATCH + b) * UNITS + u] = acc[i][r];
        }
    }
}

// ---------------------------------------------------------------------------
// Phase 2: recurrent scan. grid = 4 batch tiles, block = 512 (16 waves).
// Wave w owns N-tiles 2w, 2w+1. Per step per wave: 32 bf16 WMMAs.
// ---------------------------------------------------------------------------
__global__ void __launch_bounds__(512) rnn_scan(
        const float* __restrict__ xT, const char* __restrict__ bpack,
        const float* __restrict__ bias, const float* __restrict__ h0,
        float* __restrict__ out) {
    extern __shared__ char smem[];
    int bt   = blockIdx.x;                  // batch tile 0..3
    int tid  = threadIdx.x;
    int wave = tid >> 5, lane = tid & 31;
    int n = lane & 15, g = lane >> 4;
    int nt0 = wave * 2, nt1 = wave * 2 + 1;

    // Fill LDS with B's K-high half (chunks 8..15, all 32 N-tiles): 256KB.
    for (int s = tid; s < NTILES * 8 * 32; s += 512) {
        int ln = s & 31, cc = (s >> 5) & 7, nt = s >> 8;
        *(v16bf*)(smem + (size_t)s * 32) =
            *(const v16bf*)(bpack + (size_t)((nt * KCH_B + 8 + cc) * 32 + ln) * 32);
    }
    // Register-resident B K-low half for this wave's two N-tiles: 128 VGPRs.
    v16bf Breg0[8], Breg1[8];
#pragma unroll
    for (int c = 0; c < 8; ++c) {
        Breg0[c] = *(const v16bf*)(bpack + (size_t)((nt0 * KCH_B + c) * 32 + lane) * 32);
        Breg1[c] = *(const v16bf*)(bpack + (size_t)((nt1 * KCH_B + c) * 32 + lane) * 32);
    }
    // h state buffer 0: broadcast h0 over all 16 batch rows, A-operand layout.
    {
        int u = tid;                        // 512 threads == 512 units
        unsigned short bits = f2bf_bits(h0[u]);
        int cu = u >> 5, ku = u & 31;
        int gU = (ku >> 3) & 1;
        int jU = ((ku >> 4) << 2) | ((ku >> 1) & 3);
        int hU = ku & 1;
        char* hb0 = smem + LDS_BH_BYTES;
#pragma unroll
        for (int m = 0; m < 16; ++m) {
            int l2 = (gU << 4) | m;
            *(unsigned short*)(hb0 + (size_t)(cu * 32 + l2) * 32 + jU * 4 + hU * 2) = bits;
        }
    }
    // Per-lane constants.
    int u0 = nt0 * 16 + n, u1 = nt1 * 16 + n;
    float bias0 = bias[u0], bias1 = bias[u1];
    // Next-h scatter slots (A-layout decomposition of unit index).
    int cu0 = u0 >> 5, ku0 = u0 & 31;
    int gU0 = (ku0 >> 3) & 1, jU0 = ((ku0 >> 4) << 2) | ((ku0 >> 1) & 3), hU0 = ku0 & 1;
    int cu1 = u1 >> 5, ku1 = u1 & 31;
    int gU1 = (ku1 >> 3) & 1, jU1 = ((ku1 >> 4) << 2) | ((ku1 >> 1) & 3), hU1 = ku1 & 1;

    __syncthreads();

    for (int t = 0; t < SEQ; ++t) {
        const char* cur = smem + LDS_BH_BYTES + ((t & 1) ? LDS_H_BYTES : 0);
        char*       nxt = smem + LDS_BH_BYTES + ((t & 1) ? 0 : LDS_H_BYTES);
        size_t base = (size_t)(t * BATCH + bt * 16) * UNITS;

        // Accumulators initialized with the pre-projected input xT (fp32).
        v8f acc0 = {}, acc1 = {};
#pragma unroll
        for (int r = 0; r < 8; ++r) {
            size_t off = base + (size_t)(r + 8 * g) * UNITS;
            acc0[r] = xT[off + u0];
            acc1[r] = xT[off + u1];
        }
        if (t + 1 < SEQ) {  // uniform branch; hint next step's xT into cache
            __builtin_prefetch(&xT[base + (size_t)BATCH * UNITS +
                                   (size_t)(8 * g) * UNITS + u0], 0, 1);
        }
        // K = 0..255 from register-resident B.
#pragma unroll
        for (int c = 0; c < 8; ++c) {
            v16bf a = *(const v16bf*)(cur + (size_t)(c * 32 + lane) * 32);
            acc0 = __builtin_amdgcn_wmma_f32_16x16x32_bf16(
                false, a, false, Breg0[c], (short)0, acc0, false, false);
            acc1 = __builtin_amdgcn_wmma_f32_16x16x32_bf16(
                false, a, false, Breg1[c], (short)0, acc1, false, false);
        }
        // K = 256..511 with B operand streamed from LDS.
#pragma unroll
        for (int c = 0; c < 8; ++c) {
            v16bf a  = *(const v16bf*)(cur + (size_t)((8 + c) * 32 + lane) * 32);
            v16bf b0 = *(const v16bf*)(smem + (size_t)((nt0 * 8 + c) * 32 + lane) * 32);
            v16bf b1 = *(const v16bf*)(smem + (size_t)((nt1 * 8 + c) * 32 + lane) * 32);
            acc0 = __builtin_amdgcn_wmma_f32_16x16x32_bf16(
                false, a, false, b0, (short)0, acc0, false, false);
            acc1 = __builtin_amdgcn_wmma_f32_16x16x32_bf16(
                false, a, false, b1, (short)0, acc1, false, false);
        }
        // modrelu, emit fp32 output, scatter bf16 state for next step.
#pragma unroll
        for (int r = 0; r < 8; ++r) {
            float z0 = acc0[r], z1 = acc1[r];
            float a0 = fmaxf(fabsf(z0) + bias0, 0.0f);
            float a1 = fmaxf(fabsf(z1) + bias1, 0.0f);
            float h0v = (z0 > 0.0f) ? a0 : ((z0 < 0.0f) ? -a0 : 0.0f);
            float h1v = (z1 > 0.0f) ? a1 : ((z1 < 0.0f) ? -a1 : 0.0f);
            size_t off = base + (size_t)(r + 8 * g) * UNITS;
            out[off + u0] = h0v;
            out[off + u1] = h1v;
            int m = r + 8 * g;
            *(unsigned short*)(nxt + (size_t)(cu0 * 32 + ((gU0 << 4) | m)) * 32 +
                               jU0 * 4 + hU0 * 2) = f2bf_bits(h0v);
            *(unsigned short*)(nxt + (size_t)(cu1 * 32 + ((gU1 << 4) | m)) * 32 +
                               jU1 * 4 + hU1 * 2) = f2bf_bits(h1v);
        }
        __syncthreads();
    }
}

extern "C" void kernel_launch(void* const* d_in, const int* in_sizes, int n_in,
                              void* d_out, int out_size, void* d_ws, size_t ws_size,
                              hipStream_t stream) {
    const float* x    = (const float*)d_in[0];
    const float* Tm   = (const float*)d_in[1];
    const float* Bm   = (const float*)d_in[2];
    const float* bias = (const float*)d_in[3];
    const float* h0   = (const float*)d_in[4];
    float* out = (float*)d_out;
    char*  ws  = (char*)d_ws;

    // 288KB dynamic LDS (gfx1250 allows up to 320KB per workgroup).
    (void)hipFuncSetAttribute((const void*)rnn_scan,
                              hipFuncAttributeMaxDynamicSharedMemorySize,
                              LDS_TOTAL);

    pack_weights<<<(TPACK_SLOTS + BPACK_SLOTS) / 256, 256, 0, stream>>>(Tm, Bm, ws);
    proj_gemm<<<(BATCH * SEQ) / 16, 256, 0, stream>>>(x, ws + TPACK_OFF, (float*)ws);
    rnn_scan<<<BATCH / 16, 512, LDS_TOTAL, stream>>>(
        (const float*)ws, ws + BPACK_OFF, bias, h0, out);
}
